// SpatialTransformer_33921651704352
// MI455X (gfx1250) — matrix operand order repaired and checked by reading, not compiled, not run on hardware
//
#include <hip/hip_runtime.h>
#include <math.h>

typedef _Float16 f16_t;
typedef __attribute__((ext_vector_type(8)))  _Float16 f16x8;
typedef __attribute__((ext_vector_type(16))) _Float16 v16h;
typedef __attribute__((ext_vector_type(8)))  float    v8f;

// ---------------------------------------------------------------------------
// Batched WMMA GEMM:  D = scale * (A @ B) + bias + residual
//   A: f16 [M,K] row-major (per-(batch,head) base via strides)
//   B: f16 [N,K] row-major (weightsT / per-head K-matrix / transposed V)
// Block = 256 thr = 8 waves; block tile 128x64; wave tile 16x64 (4 WMMAs/step).
// B tile (64 n x 32 k = 4KB) staged in LDS via global_load_async_to_lds_b128,
// read back as v16h fragments with ds_load_b128.
// Epilogue modes: 0 = row-major (+optional padded-row remap)
//                 1 = BCHW scatter (+x residual) for proj_out
//                 2 = per-head transposed V store [b,h,d,token]
// ---------------------------------------------------------------------------
__global__ __launch_bounds__(256) void k_wmma_gemm(
    const f16_t* __restrict__ A, long aSB, long aSH, int lda,
    const f16_t* __restrict__ B, long bSB, long bSH, int ldb,
    float* __restrict__ outF, f16_t* __restrict__ outH, long oSB, long oSH, int ldo,
    const float* __restrict__ bias,
    const float* __restrict__ resid, long rSB, long rSH, int ldr,
    int M, int N, int K, float scale, int nh,
    int mode, int Cdim, int HWdim,
    int rowGroup, long rowPadStride,
    int tokG, int tokPad)
{
    __shared__ __attribute__((aligned(32))) f16_t btile[64 * 32];

    int z  = blockIdx.z;
    int zb = z / nh, zh = z % nh;
    const f16_t* Ab = A + (long)zb * aSB + (long)zh * aSH;
    const f16_t* Bb = B + (long)zb * bSB + (long)zh * bSH;

    int lane = threadIdx.x & 31;
    int lo = lane & 15, hi = lane >> 4;
    int m0 = (blockIdx.y * 8 + (int)(threadIdx.x >> 5)) * 16;
    int n0 = blockIdx.x * 64;

    // staging assignment: thread t -> row sn (0..63), 16B chunk sc (0..3)
    int sn = threadIdx.x >> 2;
    int sc = threadIdx.x & 3;
    int snn = n0 + sn; if (snn > N - 1) snn = N - 1;
    const f16_t* bsrc = Bb + (long)snn * ldb + sc * 8;
    unsigned lds_dst = (unsigned)(uintptr_t)&btile[sn * 32 + sc * 8];

    int arow = m0 + lo; if (arow > M - 1) arow = M - 1;
    const f16_t* ap = Ab + (long)arow * lda + hi * 16;

    v8f acc[4] = {v8f{0.f,0.f,0.f,0.f,0.f,0.f,0.f,0.f},
                  v8f{0.f,0.f,0.f,0.f,0.f,0.f,0.f,0.f},
                  v8f{0.f,0.f,0.f,0.f,0.f,0.f,0.f,0.f},
                  v8f{0.f,0.f,0.f,0.f,0.f,0.f,0.f,0.f}};

    for (int kk = 0; kk < K; kk += 32) {
        // ---- async-stage B[n0..n0+63, kk..kk+31] -> LDS (one b128 per lane) ----
        {
            unsigned long long ga = (unsigned long long)(bsrc + kk);
            asm volatile("global_load_async_to_lds_b128 %0, %1, off"
                         :: "v"(lds_dst), "v"(ga) : "memory");
            asm volatile("s_wait_asynccnt 0x0" ::: "memory");
        }
        __syncthreads();

        // ---- A fragment: two contiguous b128 loads ----
        f16x8 a0 = *(const f16x8*)(ap + kk);        // K = hi*16 + 0..7
        f16x8 a1 = *(const f16x8*)(ap + kk + 8);    // K = hi*16 + 8..15
        v16h af = __builtin_shufflevector(a0, a1,
                  0,1,2,3,4,5,6,7,8,9,10,11,12,13,14,15);
        __builtin_prefetch(ap + kk + 32, 0, 1);     // global_prefetch next A chunk

        // ---- 4 x (B fragment from LDS + WMMA) ----
        #pragma unroll
        for (int nt = 0; nt < 4; ++nt) {
            const f16_t* lp = &btile[(nt * 16 + lo) * 32 + hi * 16];
            v16h bf = *(const v16h*)lp;             // 2 x ds_load_b128
            acc[nt] = __builtin_amdgcn_wmma_f32_16x16x32_f16(
                false, af, false, bf, (short)0, acc[nt], false, false);
        }
        __syncthreads();   // protect btile before next overwrite
    }

    // ---- epilogue ----
    #pragma unroll
    for (int nt = 0; nt < 4; ++nt) {
        #pragma unroll
        for (int i = 0; i < 8; ++i) {
            int m = m0 + hi * 8 + i;
            int n = n0 + nt * 16 + lo;
            if (m >= M || n >= N) continue;
            float val = acc[nt][i] * scale;
            if (bias) val += bias[n];
            if (mode == 1) {            // BCHW scatter + residual x
                int b_ = m / HWdim, hw = m - b_ * HWdim;
                long oidx = ((long)b_ * Cdim + n) * HWdim + hw;
                if (resid) val += resid[oidx];
                if (outF) outF[oidx] = val;
                if (outH) outH[oidx] = (f16_t)val;
            } else if (mode == 2) {     // transposed-V store [b, h, d, token]
                int b_ = m / tokG, t = m - b_ * tokG;
                int h_ = n >> 6, d_ = n & 63;
                long oidx = (((long)b_ * 8 + h_) * 64 + d_) * tokPad + t;
                if (outH) outH[oidx] = (f16_t)val;
            } else {                    // row-major (+ padded row remap)
                long obase = (long)zb * oSB + (long)zh * oSH;
                long orow;
                if (rowGroup > 0) {
                    int g = m / rowGroup, t = m - g * rowGroup;
                    orow = (long)g * rowPadStride + (long)t * ldo;
                } else {
                    orow = (long)m * ldo;
                }
                if (resid) val += resid[(long)zb * rSB + (long)zh * rSH + (long)m * ldr + n];
                long oidx = obase + orow + n;
                if (outF) outF[oidx] = val;
                if (outH) outH[oidx] = (f16_t)val;
            }
        }
    }
}

// ---------------------------------------------------------------------------
// GroupNorm over [B,C,H,W] (32 groups), writes f16 [B, HW, C]
// ---------------------------------------------------------------------------
__global__ __launch_bounds__(256) void k_groupnorm(
    const float* __restrict__ x, const float* __restrict__ g, const float* __restrict__ b,
    f16_t* __restrict__ outH, int Cc, int HW, int cpg)
{
    int grp = blockIdx.x, bb = blockIdx.y;
    const float* xp = x + ((long)bb * Cc + (long)grp * cpg) * HW;
    int n = cpg * HW;
    float s = 0.f, ss = 0.f;
    for (int i = threadIdx.x; i < n; i += blockDim.x) {
        float v = xp[i]; s += v; ss += v * v;
    }
    __shared__ float sh1[256], sh2[256];
    sh1[threadIdx.x] = s; sh2[threadIdx.x] = ss;
    __syncthreads();
    for (int off = 128; off > 0; off >>= 1) {
        if ((int)threadIdx.x < off) {
            sh1[threadIdx.x] += sh1[threadIdx.x + off];
            sh2[threadIdx.x] += sh2[threadIdx.x + off];
        }
        __syncthreads();
    }
    float mu  = sh1[0] / n;
    float var = sh2[0] / n - mu * mu;
    float inv = rsqrtf(var + 1e-6f);
    for (int i = threadIdx.x; i < n; i += blockDim.x) {
        int c = grp * cpg + i / HW;
        int hw = i % HW;
        float v = (xp[i] - mu) * inv * g[c] + b[c];
        outH[((long)bb * HW + hw) * Cc + c] = (f16_t)v;
    }
}

// ---------------------------------------------------------------------------
// LayerNorm over rows of [rows, Cc] fp32 -> f16
// ---------------------------------------------------------------------------
__global__ __launch_bounds__(256) void k_layernorm(
    const float* __restrict__ x, const float* __restrict__ g, const float* __restrict__ b,
    f16_t* __restrict__ outH, int Cc)
{
    long row = blockIdx.x;
    const float* xp = x + row * Cc;
    float s = 0.f, ss = 0.f;
    for (int i = threadIdx.x; i < Cc; i += blockDim.x) {
        float v = xp[i]; s += v; ss += v * v;
    }
    __shared__ float sh1[256], sh2[256];
    sh1[threadIdx.x] = s; sh2[threadIdx.x] = ss;
    __syncthreads();
    for (int off = 128; off > 0; off >>= 1) {
        if ((int)threadIdx.x < off) {
            sh1[threadIdx.x] += sh1[threadIdx.x + off];
            sh2[threadIdx.x] += sh2[threadIdx.x + off];
        }
        __syncthreads();
    }
    float mu  = sh1[0] / Cc;
    float var = sh2[0] / Cc - mu * mu;
    float inv = rsqrtf(var + 1e-5f);
    for (int i = threadIdx.x; i < Cc; i += blockDim.x)
        outH[row * Cc + i] = (f16_t)((xp[i] - mu) * inv * g[i] + b[i]);
}

// ---------------------------------------------------------------------------
// Row softmax: fp32 [rows, Lpad] (L valid) -> f16 probs, zero-padded to Lpad
// ---------------------------------------------------------------------------
__global__ __launch_bounds__(256) void k_softmax(
    const float* __restrict__ s, f16_t* __restrict__ p, int Lpad, int L)
{
    long row = blockIdx.x;
    const float* sp = s + row * Lpad;
    f16_t* pp = p + row * Lpad;
    float mx = -3.0e38f;
    for (int i = threadIdx.x; i < L; i += blockDim.x) mx = fmaxf(mx, sp[i]);
    __shared__ float sh[256];
    sh[threadIdx.x] = mx; __syncthreads();
    for (int off = 128; off > 0; off >>= 1) {
        if ((int)threadIdx.x < off) sh[threadIdx.x] = fmaxf(sh[threadIdx.x], sh[threadIdx.x + off]);
        __syncthreads();
    }
    mx = sh[0]; __syncthreads();
    float sum = 0.f;
    for (int i = threadIdx.x; i < L; i += blockDim.x) sum += __expf(sp[i] - mx);
    sh[threadIdx.x] = sum; __syncthreads();
    for (int off = 128; off > 0; off >>= 1) {
        if ((int)threadIdx.x < off) sh[threadIdx.x] += sh[threadIdx.x + off];
        __syncthreads();
    }
    float inv = 1.f / sh[0];
    for (int i = threadIdx.x; i < Lpad; i += blockDim.x)
        pp[i] = (f16_t)(i < L ? __expf(sp[i] - mx) * inv : 0.f);
}

// ---------------------------------------------------------------------------
// GeGLU: z fp32 [M, 2*DFF] -> f16 [M, DFF] : val * gelu_exact(gate)
// ---------------------------------------------------------------------------
__global__ __launch_bounds__(256) void k_geglu(
    const float* __restrict__ z, f16_t* __restrict__ out, long M, int DFFd)
{
    long idx = (long)blockIdx.x * blockDim.x + threadIdx.x;
    if (idx >= M * DFFd) return;
    long m = idx / DFFd; int n = (int)(idx - m * DFFd);
    float val  = z[m * 2 * DFFd + n];
    float gate = z[m * 2 * DFFd + DFFd + n];
    float ge = 0.5f * gate * (1.f + erff(gate * 0.70710678118654752f));
    out[idx] = (f16_t)(val * ge);
}

// f32 [K,N] weight -> f16 [N,K] transposed
__global__ __launch_bounds__(256) void k_wt(const float* __restrict__ w,
                                            f16_t* __restrict__ o, int Kd, int Nd)
{
    long idx = (long)blockIdx.x * blockDim.x + threadIdx.x;
    if (idx >= (long)Kd * Nd) return;
    int k = (int)(idx / Nd), n = (int)(idx - (long)k * Nd);
    o[(long)n * Kd + k] = (f16_t)w[idx];
}

__global__ __launch_bounds__(256) void k_cvt(const float* __restrict__ x,
                                             f16_t* __restrict__ o, long n)
{
    long i = (long)blockIdx.x * blockDim.x + threadIdx.x;
    if (i < n) o[i] = (f16_t)x[i];
}

__global__ __launch_bounds__(256) void k_zero16(f16_t* __restrict__ o, long n)
{
    long i = (long)blockIdx.x * blockDim.x + threadIdx.x;
    if (i < n) o[i] = (f16_t)0.f;
}

// ---------------------------------------------------------------------------
// Host orchestration
// ---------------------------------------------------------------------------
static void gemm(hipStream_t st,
                 const f16_t* A, long aSB, long aSH, int lda,
                 const f16_t* B, long bSB, long bSH, int ldb,
                 float* outF, f16_t* outH, long oSB, long oSH, int ldo,
                 const float* bias, const float* resid, long rSB, long rSH, int ldr,
                 int M, int N, int K, float scale, int batches, int nh,
                 int mode = 0, int Cdim = 0, int HWdim = 0,
                 int rowGroup = 0, long rowPadStride = 0,
                 int tokG = 0, int tokPad = 0)
{
    dim3 grid((N + 63) / 64, (M + 127) / 128, batches);
    k_wmma_gemm<<<grid, 256, 0, st>>>(A, aSB, aSH, lda, B, bSB, bSH, ldb,
                                      outF, outH, oSB, oSH, ldo, bias,
                                      resid, rSB, rSH, ldr, M, N, K, scale, nh,
                                      mode, Cdim, HWdim, rowGroup, rowPadStride,
                                      tokG, tokPad);
}

extern "C" void kernel_launch(void* const* d_in, const int* in_sizes, int n_in,
                              void* d_out, int out_size, void* d_ws, size_t ws_size,
                              hipStream_t stream)
{
    (void)in_sizes; (void)n_in; (void)out_size; (void)ws_size;
    const int Bn = 4, Cc = 512, HWn = 1024, NHn = 8, DHn = 64;
    const int NCn = 77, NCp = 96, DCn = 768, DFFn = 2048;
    const int Mn = Bn * HWn;                 // 4096
    const int Sn = HWn;                      // 1024 query tokens / batch

    const float* x    = (const float*)d_in[0];
    const float* cond = (const float*)d_in[1];
    const float* gn_g = (const float*)d_in[2];
    const float* gn_b = (const float*)d_in[3];
    const float* w_pi = (const float*)d_in[4];
    const float* b_pi = (const float*)d_in[5];
    const float* w_po = (const float*)d_in[6];
    const float* b_po = (const float*)d_in[7];
    const float* ln1g = (const float*)d_in[8];
    const float* ln1b = (const float*)d_in[9];
    const float* ln2g = (const float*)d_in[10];
    const float* ln2b = (const float*)d_in[11];
    const float* ln3g = (const float*)d_in[12];
    const float* ln3b = (const float*)d_in[13];
    const float* a1wq = (const float*)d_in[14];
    const float* a1wk = (const float*)d_in[15];
    const float* a1wv = (const float*)d_in[16];
    const float* a1wo = (const float*)d_in[17];
    const float* a1bo = (const float*)d_in[18];
    const float* a2wq = (const float*)d_in[19];
    const float* a2wk = (const float*)d_in[20];
    const float* a2wv = (const float*)d_in[21];
    const float* a2wo = (const float*)d_in[22];
    const float* a2bo = (const float*)d_in[23];
    const float* ffw1 = (const float*)d_in[24];
    const float* ffb1 = (const float*)d_in[25];
    const float* ffw2 = (const float*)d_in[26];
    const float* ffb2 = (const float*)d_in[27];
    float* out        = (float*)d_out;

    // bump allocator over d_ws
    size_t off = 0;
    auto alloc = [&](size_t bytes) -> void* {
        off = (off + 255) & ~(size_t)255;
        void* r = (char*)d_ws + off;
        off += bytes;
        return r;
    };
    auto h = [&](size_t elems) { return (f16_t*)alloc(elems * sizeof(f16_t)); };
    auto f = [&](size_t elems) { return (float*)alloc(elems * sizeof(float)); };

    f16_t* wt_pi  = h(512 * 512);
    f16_t* wt_q1  = h(512 * 512);
    f16_t* wt_k1  = h(512 * 512);
    f16_t* wt_v1  = h(512 * 512);
    f16_t* wt_o1  = h(512 * 512);
    f16_t* wt_q2  = h(512 * 512);
    f16_t* wt_k2  = h((size_t)512 * 768);
    f16_t* wt_v2  = h((size_t)512 * 768);
    f16_t* wt_o2  = h(512 * 512);
    f16_t* wt_f1  = h((size_t)4096 * 512);
    f16_t* wt_f2  = h((size_t)512 * 2048);
    f16_t* wt_po  = h(512 * 512);
    f16_t* cond_h = h((size_t)Bn * NCn * DCn);
    f16_t* kc_h   = h((size_t)Bn * NCp * Cc);          // padded K rows [B,96,C]
    f16_t* vt_c   = h((size_t)Bn * NHn * DHn * NCp);   // cross V^T [b,h,d,96]
    f16_t* a_in   = h((size_t)Mn * Cc);
    f16_t* lnh    = h((size_t)Mn * Cc);
    f16_t* q_h    = h((size_t)Mn * Cc);
    f16_t* k_h    = h((size_t)Mn * Cc);
    f16_t* vt_s   = h((size_t)Bn * NHn * DHn * Sn);    // self V^T [b,h,d,1024]
    f16_t* attn_h = h((size_t)Mn * Cc);
    f16_t* ffh    = h((size_t)Mn * DFFn);
    f16_t* y3h    = h((size_t)Mn * Cc);
    f16_t* probs  = h((size_t)Bn * NHn * Sn * 1024);
    float* y0     = f((size_t)Mn * Cc);
    float* y1     = f((size_t)Mn * Cc);
    float* y2     = f((size_t)Mn * Cc);
    float* scores = f((size_t)Bn * NHn * Sn * 1024);   // reused as FFN z [4096,4096]

    auto blocks = [](long n) { return (unsigned)((n + 255) / 256); };

    // --- weight conversions (f32 [K,N] -> f16 [N,K]) ---
    k_wt<<<blocks(512 * 512), 256, 0, stream>>>(w_pi, wt_pi, 512, 512);
    k_wt<<<blocks(512 * 512), 256, 0, stream>>>(a1wq, wt_q1, 512, 512);
    k_wt<<<blocks(512 * 512), 256, 0, stream>>>(a1wk, wt_k1, 512, 512);
    k_wt<<<blocks(512 * 512), 256, 0, stream>>>(a1wv, wt_v1, 512, 512);
    k_wt<<<blocks(512 * 512), 256, 0, stream>>>(a1wo, wt_o1, 512, 512);
    k_wt<<<blocks(512 * 512), 256, 0, stream>>>(a2wq, wt_q2, 512, 512);
    k_wt<<<blocks((long)768 * 512), 256, 0, stream>>>(a2wk, wt_k2, 768, 512);
    k_wt<<<blocks((long)768 * 512), 256, 0, stream>>>(a2wv, wt_v2, 768, 512);
    k_wt<<<blocks(512 * 512), 256, 0, stream>>>(a2wo, wt_o2, 512, 512);
    k_wt<<<blocks((long)512 * 4096), 256, 0, stream>>>(ffw1, wt_f1, 512, 4096);
    k_wt<<<blocks((long)2048 * 512), 256, 0, stream>>>(ffw2, wt_f2, 2048, 512);
    k_wt<<<blocks(512 * 512), 256, 0, stream>>>(w_po, wt_po, 512, 512);
    k_cvt<<<blocks((long)Bn * NCn * DCn), 256, 0, stream>>>(cond, cond_h, (long)Bn * NCn * DCn);
    k_zero16<<<blocks((long)Bn * NCp * Cc), 256, 0, stream>>>(kc_h, (long)Bn * NCp * Cc);
    k_zero16<<<blocks((long)Bn * NHn * DHn * NCp), 256, 0, stream>>>(vt_c, (long)Bn * NHn * DHn * NCp);

    // --- GroupNorm -> a_in [B*HW, C] f16 ---
    k_groupnorm<<<dim3(32, Bn), 256, 0, stream>>>(x, gn_g, gn_b, a_in, Cc, HWn, Cc / 32);

    // --- proj_in: y0 = a_in @ Wpi + b ---
    gemm(stream, a_in, 0, 0, Cc, wt_pi, 0, 0, Cc,
         y0, nullptr, 0, 0, Cc, b_pi, nullptr, 0, 0, 0,
         Mn, Cc, Cc, 1.f, 1, 1);

    // ===================== self-attention =====================
    k_layernorm<<<Mn, 256, 0, stream>>>(y0, ln1g, ln1b, lnh, Cc);
    gemm(stream, lnh, 0, 0, Cc, wt_q1, 0, 0, Cc, nullptr, q_h, 0, 0, Cc,
         nullptr, nullptr, 0, 0, 0, Mn, Cc, Cc, 1.f, 1, 1);
    gemm(stream, lnh, 0, 0, Cc, wt_k1, 0, 0, Cc, nullptr, k_h, 0, 0, Cc,
         nullptr, nullptr, 0, 0, 0, Mn, Cc, Cc, 1.f, 1, 1);
    // V projection with per-head transposed store [b,h,d,1024]
    gemm(stream, lnh, 0, 0, Cc, wt_v1, 0, 0, Cc, nullptr, vt_s, 0, 0, Cc,
         nullptr, nullptr, 0, 0, 0, Mn, Cc, Cc, 1.f, 1, 1,
         /*mode=*/2, 0, 0, 0, 0, /*tokG=*/Sn, /*tokPad=*/Sn);

    // scores[b,h,i,j] = scale * q.k  (32 batch-heads)
    gemm(stream, q_h, (long)Sn * Cc, DHn, Cc,
         k_h, (long)Sn * Cc, DHn, Cc,
         scores, nullptr, (long)NHn * Sn * 1024, (long)Sn * 1024, 1024,
         nullptr, nullptr, 0, 0, 0,
         Sn, Sn, DHn, 0.125f, Bn * NHn, NHn);
    k_softmax<<<Bn * NHn * Sn, 256, 0, stream>>>(scores, probs, 1024, 1024);

    // attn_h[b,i,h*64+d] = probs @ V   (B = V^T rows d, contiguous over tokens)
    gemm(stream, probs, (long)NHn * Sn * 1024, (long)Sn * 1024, 1024,
         vt_s, (long)NHn * DHn * Sn, (long)DHn * Sn, Sn,
         nullptr, attn_h, (long)Sn * Cc, DHn, Cc,
         nullptr, nullptr, 0, 0, 0,
         Sn, DHn, Sn, 1.f, Bn * NHn, NHn);

    // out-proj + residual -> y1
    gemm(stream, attn_h, 0, 0, Cc, wt_o1, 0, 0, Cc,
         y1, nullptr, 0, 0, Cc, a1bo, y0, 0, 0, Cc,
         Mn, Cc, Cc, 1.f, 1, 1);

    // ===================== cross-attention =====================
    k_layernorm<<<Mn, 256, 0, stream>>>(y1, ln2g, ln2b, lnh, Cc);
    gemm(stream, lnh, 0, 0, Cc, wt_q2, 0, 0, Cc, nullptr, q_h, 0, 0, Cc,
         nullptr, nullptr, 0, 0, 0, Mn, Cc, Cc, 1.f, 1, 1);
    // K from cond -> padded [B, 96, C]
    gemm(stream, cond_h, 0, 0, DCn, wt_k2, 0, 0, DCn,
         nullptr, kc_h, 0, 0, Cc, nullptr, nullptr, 0, 0, 0,
         Bn * NCn, Cc, DCn, 1.f, 1, 1, 0, 0, 0, NCn, (long)NCp * Cc);
    // V from cond -> transposed [b,h,d,96] (pre-zeroed padding)
    gemm(stream, cond_h, 0, 0, DCn, wt_v2, 0, 0, DCn,
         nullptr, vt_c, 0, 0, Cc, nullptr, nullptr, 0, 0, 0,
         Bn * NCn, Cc, DCn, 1.f, 1, 1,
         /*mode=*/2, 0, 0, 0, 0, /*tokG=*/NCn, /*tokPad=*/NCp);

    // cross scores [32, 1024, 96] (77 valid)
    gemm(stream, q_h, (long)Sn * Cc, DHn, Cc,
         kc_h, (long)NCp * Cc, DHn, Cc,
         scores, nullptr, (long)NHn * Sn * NCp, (long)Sn * NCp, NCp,
         nullptr, nullptr, 0, 0, 0,
         Sn, NCn, DHn, 0.125f, Bn * NHn, NHn);
    k_softmax<<<Bn * NHn * Sn, 256, 0, stream>>>(scores, probs, NCp, NCn);

    gemm(stream, probs, (long)NHn * Sn * NCp, (long)Sn * NCp, NCp,
         vt_c, (long)NHn * DHn * NCp, (long)DHn * NCp, NCp,
         nullptr, attn_h, (long)Sn * Cc, DHn, Cc,
         nullptr, nullptr, 0, 0, 0,
         Sn, DHn, NCp, 1.f, Bn * NHn, NHn);

    gemm(stream, attn_h, 0, 0, Cc, wt_o2, 0, 0, Cc,
         y2, nullptr, 0, 0, Cc, a2bo, y1, 0, 0, Cc,
         Mn, Cc, Cc, 1.f, 1, 1);

    // ===================== GeGLU FFN =====================
    k_layernorm<<<Mn, 256, 0, stream>>>(y2, ln3g, ln3b, lnh, Cc);
    gemm(stream, lnh, 0, 0, Cc, wt_f1, 0, 0, Cc,
         scores, nullptr, 0, 0, 2 * DFFn, ffb1, nullptr, 0, 0, 0,
         Mn, 2 * DFFn, Cc, 1.f, 1, 1);
    k_geglu<<<blocks((long)Mn * DFFn), 256, 0, stream>>>(scores, ffh, Mn, DFFn);
    gemm(stream, ffh, 0, 0, DFFn, wt_f2, 0, 0, DFFn,
         nullptr, y3h, 0, 0, Cc, ffb2, y2, 0, 0, Cc,
         Mn, Cc, DFFn, 1.f, 1, 1);

    // ===================== proj_out + input residual (BCHW scatter) =====================
    gemm(stream, y3h, 0, 0, Cc, wt_po, 0, 0, Cc,
         out, nullptr, 0, 0, Cc, b_po, x, 0, 0, 0,
         Mn, Cc, Cc, 1.f, 1, 1, /*mode=*/1, Cc, HWn);
}
